// NAT_19138374271496
// MI455X (gfx1250) — compile-verified, matrix operand order checked by reading, not run on hardware
//
#include <hip/hip_runtime.h>
#include <math.h>

// ---------------------------------------------------------------------------
// NAT forward for MI455X (gfx1250, wave32).
// Weight GEMMs: block-cooperative WMMA (128x64 block tile, 32x64 per wave)
// with async global->LDS staging of the B panel (double buffered, ASYNCcnt).
// Attention GEMMs: direct-global 16x64 wave tiles.  All operands bf16
// fragment-ready, fp32 accumulation.
// ---------------------------------------------------------------------------

typedef __attribute__((ext_vector_type(16))) __bf16 v16bf;
typedef __attribute__((ext_vector_type(8)))  __bf16 bf8;
typedef __attribute__((ext_vector_type(8)))  float  v8f;

#define E_DIM 512
#define NB    8
#define SEQ_S 64
#define SEQ_T 512
#define NLAY  6
#define LDSTRIDE 40   // padded LDS row stride (bf16 elems) = 80B

__device__ __forceinline__ __bf16 f2bf(float f) { return (__bf16)f; }

__device__ __forceinline__ float pe_val(int pos, int c) {
  int i2 = c & ~1;
  float div = expf(-(float)i2 * (9.210340371976184f / 512.0f)); // ln(10000)/E
  float ang = (float)pos * div;
  return (c & 1) ? cosf(ang) : sinf(ang);
}

// CDNA5 async path: copy 16B global -> LDS, tracked by ASYNCcnt.
__device__ __forceinline__ void async_g2l_b128(unsigned lds_off, const void* g)
{
  unsigned long long ga = (unsigned long long)g;
  asm volatile("global_load_async_to_lds_b128 %0, %1, off"
               :: "v"(lds_off), "v"(ga) : "memory");
}
__device__ __forceinline__ void wait_asynccnt0()
{
  asm volatile("s_wait_asynccnt 0" ::: "memory");
}

// ---------------------------------------------------------------------------
// Block-cooperative weight GEMM:  C = relu?(A @ B^T + bias)
// Requires M%128==0, Nc%64==0, Kd%32==0.
// ---------------------------------------------------------------------------
template <int OUTBF>
__global__ void k_gemm_big(const __bf16* __restrict__ A, long lda,
                           const __bf16* __restrict__ B, long ldb,
                           const float* __restrict__ bias,
                           void* __restrict__ Cv, long ldc,
                           int M, int Nc, int Kd, int relu, float scale)
{
  __shared__ unsigned short shB[2][64 * LDSTRIDE];

  const int tn4  = Nc >> 6;
  const int m0b  = (blockIdx.x / tn4) << 7;
  const int n0   = (blockIdx.x % tn4) << 6;
  const int wave = threadIdx.x >> 5;
  const int m0   = m0b + (wave << 5);
  const int lane = threadIdx.x & 31;
  const int l    = lane & 15;
  const int hi   = lane >> 4;
  const int tid  = threadIdx.x;

  // staging assignment: 256 16B chunks (64 cols x 4 k-chunks), 2 per thread
  const int col0 = tid >> 2,         q0 = tid & 3;
  const int col1 = (tid + 128) >> 2, q1 = tid & 3;   // +128 -> +32 cols, same q
  const unsigned lo0 = (unsigned)(col0 * LDSTRIDE + q0 * 8) * 2;
  const unsigned lo1 = (unsigned)(col1 * LDSTRIDE + q1 * 8) * 2;
  const __bf16* g0 = B + (long)(n0 + col0) * ldb + q0 * 8;
  const __bf16* g1 = B + (long)(n0 + col1) * ldb + q1 * 8;

  const __bf16* Ar0 = A + (long)(m0 + l) * lda;
  const __bf16* Ar1 = Ar0 + 16 * lda;

  v8f acc[2][4] = {{{}, {}, {}, {}}, {{}, {}, {}, {}}};

  // prologue: stage k0 = 0 into buffer 0
  {
    unsigned base = (unsigned)(size_t)&shB[0][0];
    async_g2l_b128(base + lo0, g0);
    async_g2l_b128(base + lo1, g1);
  }

  for (int k0 = 0; k0 < Kd; k0 += 32) {
    const int cur = (k0 >> 5) & 1;
    wait_asynccnt0();
    __syncthreads();

    if (k0 + 32 < Kd) {
      unsigned base = (unsigned)(size_t)&shB[cur ^ 1][0];
      async_g2l_b128(base + lo0, g0 + k0 + 32);
      async_g2l_b128(base + lo1, g1 + k0 + 32);
    }

    v16bf af0, af1;
    {
      const __bf16* p0 = Ar0 + k0 + hi * 8;
      const __bf16* p1 = Ar1 + k0 + hi * 8;
      bf8 x0 = *(const bf8*)(p0), x1 = *(const bf8*)(p0 + 16);
      bf8 y0 = *(const bf8*)(p1), y1 = *(const bf8*)(p1 + 16);
#pragma unroll
      for (int e = 0; e < 8; ++e) {
        af0[e] = x0[e]; af0[e + 8] = x1[e];
        af1[e] = y0[e]; af1[e + 8] = y1[e];
      }
    }

    const unsigned short* bb = &shB[cur][0];
#pragma unroll
    for (int j = 0; j < 4; ++j) {
      const unsigned short* bp = bb + (j * 16 + l) * LDSTRIDE + hi * 8;
      bf8 b0 = *(const bf8*)(bp);
      bf8 b1 = *(const bf8*)(bp + 16);
      v16bf bv;
#pragma unroll
      for (int e = 0; e < 8; ++e) { bv[e] = b0[e]; bv[e + 8] = b1[e]; }
      acc[0][j] = __builtin_amdgcn_wmma_f32_16x16x32_bf16(false, af0, false, bv,
                                                          (short)0, acc[0][j], false, false);
      acc[1][j] = __builtin_amdgcn_wmma_f32_16x16x32_bf16(false, af1, false, bv,
                                                          (short)0, acc[1][j], false, false);
    }
    __syncthreads();
  }

#pragma unroll
  for (int mi = 0; mi < 2; ++mi) {
#pragma unroll
    for (int j = 0; j < 4; ++j) {
      const int bcol = n0 + j * 16 + l;
      const float bv = bias ? bias[bcol] : 0.0f;
#pragma unroll
      for (int r = 0; r < 8; ++r) {
        int row = m0 + mi * 16 + hi * 8 + r;
        float v = acc[mi][j][r] * scale + bv;
        if (relu) v = fmaxf(v, 0.0f);
        if (OUTBF) ((__bf16*)Cv)[(long)row * ldc + bcol] = f2bf(v);
        else       ((float*) Cv)[(long)row * ldc + bcol] = v;
      }
    }
  }
}

// ---------------------------------------------------------------------------
// Batched direct-global GEMM for attention (16x64 per wave), bf16 in.
// ---------------------------------------------------------------------------
template <int OUTBF>
__global__ void k_gemm(const __bf16* __restrict__ A, long lda, long sA,
                       const __bf16* __restrict__ B, long ldb, long sB,
                       const float* __restrict__ bias,
                       void* __restrict__ Cv, long ldc, long sC,
                       int M, int Nc, int Kd, int relu, float scale)
{
  const int tn4 = Nc >> 6;
  const int tm  = M >> 4;
  const int wid = blockIdx.x * (blockDim.x >> 5) + (threadIdx.x >> 5);
  if (wid >= tm * tn4) return;
  const int m0   = (wid / tn4) << 4;
  const int n0   = (wid % tn4) << 6;
  const int lane = threadIdx.x & 31;
  const int l    = lane & 15;
  const int hi   = lane >> 4;

  A += (long)blockIdx.z * sA;
  B += (long)blockIdx.z * sB;

  const __bf16* Arow = A + (long)(m0 + l) * lda;

  v8f acc[4] = {{}, {}, {}, {}};

  for (int k0 = 0; k0 < Kd; k0 += 32) {
    const __bf16* ap = Arow + k0 + hi * 8;
    bf8 a0 = *(const bf8*)(ap);
    bf8 a1 = *(const bf8*)(ap + 16);
    bf8 br[4][2];
#pragma unroll
    for (int j = 0; j < 4; ++j) {
      const __bf16* bp = B + (long)(n0 + j * 16 + l) * ldb + k0 + hi * 8;
      br[j][0] = *(const bf8*)(bp);
      br[j][1] = *(const bf8*)(bp + 16);
    }
    v16bf afv;
#pragma unroll
    for (int e = 0; e < 8; ++e) { afv[e] = a0[e]; afv[e + 8] = a1[e]; }
#pragma unroll
    for (int j = 0; j < 4; ++j) {
      v16bf bv;
#pragma unroll
      for (int e = 0; e < 8; ++e) { bv[e] = br[j][0][e]; bv[e + 8] = br[j][1][e]; }
      acc[j] = __builtin_amdgcn_wmma_f32_16x16x32_bf16(false, afv, false, bv,
                                                       (short)0, acc[j], false, false);
    }
  }

#pragma unroll
  for (int j = 0; j < 4; ++j) {
    const int bcol = n0 + j * 16 + l;
    const float bb = bias ? bias[bcol] : 0.0f;
#pragma unroll
    for (int r = 0; r < 8; ++r) {
      int row = m0 + hi * 8 + r;
      float v = acc[j][r] * scale + bb;
      if (relu) v = fmaxf(v, 0.0f);
      if (OUTBF) ((__bf16*)Cv)[(long)blockIdx.z * sC + (long)row * ldc + bcol] = f2bf(v);
      else       ((float*) Cv)[(long)blockIdx.z * sC + (long)row * ldc + bcol] = v;
    }
  }
}

// Weight convert+transpose: Bt[n][k] = bf16(B[k][n]).
__global__ void k_wconv(const float* __restrict__ B, __bf16* __restrict__ Bt,
                        int Kd, int Nc)
{
  __shared__ float tile[32][33];
  const int k0 = blockIdx.x * 32, n0 = blockIdx.y * 32;
  const int tx = threadIdx.x & 31, ty = threadIdx.x >> 5;
#pragma unroll
  for (int i = ty; i < 32; i += 8)
    tile[i][tx] = B[(long)(k0 + i) * Nc + n0 + tx];
  __syncthreads();
#pragma unroll
  for (int i = ty; i < 32; i += 8)
    Bt[(long)(n0 + i) * Kd + k0 + tx] = f2bf(tile[tx][i]);
}

// Per-(n,h) V transpose (bf16): Vt[z][d][k] = Vb[(k*NB+n)*E + h*64 + d].
__global__ void k_vtrans(const __bf16* __restrict__ Vb, __bf16* __restrict__ Vt, int Lk)
{
  __shared__ unsigned short sh[32][33];
  const int k0 = blockIdx.x * 32, d0 = blockIdx.y * 32, z = blockIdx.z;
  const int tx = threadIdx.x & 31, ty = threadIdx.x >> 5;
  const unsigned short* src = (const unsigned short*)Vb;
  unsigned short*       dst = (unsigned short*)Vt;
#pragma unroll
  for (int i = ty; i < 32; i += 8)
    sh[i][tx] = src[(long)(k0 + i) * 4096 + z * 64 + d0 + tx];
  __syncthreads();
#pragma unroll
  for (int i = ty; i < 32; i += 8)
    dst[(long)z * 64 * Lk + (long)(d0 + i) * Lk + k0 + tx] = sh[tx][i];
}

__global__ void k_embed(const int* __restrict__ inp, const float* __restrict__ emb,
                        float* __restrict__ inputE, float* __restrict__ x0,
                        __bf16* __restrict__ xb)
{
  int row = blockIdx.x;
  int s = row / NB, n = row % NB;
  int tok = inp[n * SEQ_S + s];
  const float* e = emb + (long)tok * E_DIM;
  float*  pi = inputE + (long)row * E_DIM;
  float*  px = x0     + (long)row * E_DIM;
  __bf16* pb = xb     + (long)row * E_DIM;
  for (int c = threadIdx.x; c < E_DIM; c += blockDim.x) {
    float v = e[c];
    float o = v + pe_val(s, c);
    pi[c] = v; px[c] = o; pb[c] = f2bf(o);
  }
}

__global__ void k_softmax_bf(__bf16* __restrict__ S, int cols)
{
  __bf16* p = S + (size_t)blockIdx.x * cols;
  __shared__ float red[256];
  int t = threadIdx.x;
  float m = -1e30f;
  for (int c = t; c < cols; c += blockDim.x) m = fmaxf(m, (float)p[c]);
  red[t] = m; __syncthreads();
  for (int s = 128; s > 0; s >>= 1) {
    if (t < s) red[t] = fmaxf(red[t], red[t + s]);
    __syncthreads();
  }
  m = red[0]; __syncthreads();
  float sum = 0.0f;
  for (int c = t; c < cols; c += blockDim.x) sum += expf((float)p[c] - m);
  red[t] = sum; __syncthreads();
  for (int s = 128; s > 0; s >>= 1) {
    if (t < s) red[t] += red[t + s];
    __syncthreads();
  }
  float inv = 1.0f / red[0];
  for (int c = t; c < cols; c += blockDim.x)
    p[c] = f2bf(expf((float)p[c] - m) * inv);
}

__global__ void k_add_ln(const float* __restrict__ X, const float* __restrict__ D,
                         const float* __restrict__ g, const float* __restrict__ b,
                         float* __restrict__ O, __bf16* __restrict__ Ob)
{
  long row = blockIdx.x;
  const float* x = X + row * E_DIM;
  const float* d = D + row * E_DIM;
  float*       o = O + row * E_DIM;
  __bf16*      ob = Ob + row * E_DIM;
  int t = threadIdx.x;
  float v0 = x[t] + d[t];
  float v1 = x[t + 256] + d[t + 256];
  __shared__ float red[256];
  red[t] = v0 + v1; __syncthreads();
  for (int s = 128; s > 0; s >>= 1) { if (t < s) red[t] += red[t + s]; __syncthreads(); }
  float mean = red[0] * (1.0f / 512.0f); __syncthreads();
  float a0 = v0 - mean, a1 = v1 - mean;
  red[t] = a0 * a0 + a1 * a1; __syncthreads();
  for (int s = 128; s > 0; s >>= 1) { if (t < s) red[t] += red[t + s]; __syncthreads(); }
  float inv = rsqrtf(red[0] * (1.0f / 512.0f) + 1e-5f);
  float r0 = a0 * inv * g[t]       + b[t];
  float r1 = a1 * inv * g[t + 256] + b[t + 256];
  o[t] = r0; o[t + 256] = r1;
  ob[t] = f2bf(r0); ob[t + 256] = f2bf(r1);
}

__global__ void k_fert_logits(const float* __restrict__ X, const float* __restrict__ Wf,
                              const float* __restrict__ bfert, float* __restrict__ L)
{
  int row = blockIdx.x;
  const float* x = X + (long)row * E_DIM;
  __shared__ float red[256];
  int t = threadIdx.x;
  int col = t & 7;
  int ks  = t >> 3;
  float s = 0.0f;
#pragma unroll
  for (int j = 0; j < 16; ++j) {
    int k = ks * 16 + j;
    s += x[k] * Wf[(long)k * 8 + col];
  }
  red[t] = s; __syncthreads();
  for (int off = 128; off >= 8; off >>= 1) {
    if (t < off) red[t] += red[t + off];
    __syncthreads();
  }
  if (t < 8) L[(long)row * 8 + t] = red[t] + bfert[t];
}

__global__ void k_argmax(const float* __restrict__ logits, int* __restrict__ fert)
{
  int i = blockIdx.x * blockDim.x + threadIdx.x;
  if (i >= SEQ_S * NB) return;
  const float* p = logits + (long)i * 8;
  int best = 0; float bv = p[0];
  for (int j = 1; j < 8; ++j) { float v = p[j]; if (v > bv) { bv = v; best = j; } }
  fert[i] = best;
}

__global__ void k_cumsum(const int* __restrict__ fert, int* __restrict__ cum)
{
  __shared__ int sh[SEQ_S];
  int n = blockIdx.x, s = threadIdx.x;
  sh[s] = fert[s * NB + n];
  __syncthreads();
  for (int off = 1; off < SEQ_S; off <<= 1) {
    int v = (s >= off) ? sh[s - off] : 0;
    __syncthreads();
    sh[s] += v;
    __syncthreads();
  }
  cum[s * NB + n] = sh[s];
}

__global__ void k_copyfert(const float* __restrict__ inputE, const int* __restrict__ cum,
                           float* __restrict__ y, __bf16* __restrict__ yb)
{
  int row = blockIdx.x;
  int t = row / NB, n = row % NB;
  __shared__ int s_idx, s_valid;
  if (threadIdx.x == 0) {
    int idx = 0;
    for (int s = 0; s < SEQ_S; ++s) idx += (cum[s * NB + n] <= t) ? 1 : 0;
    int total = cum[(SEQ_S - 1) * NB + n];
    s_idx   = idx < (SEQ_S - 1) ? idx : (SEQ_S - 1);
    s_valid = (t < total) ? 1 : 0;
  }
  __syncthreads();
  int idx = s_idx, valid = s_valid;
  const float* src = inputE + ((long)idx * NB + n) * E_DIM;
  float*  dst  = y  + (long)row * E_DIM;
  __bf16* dstb = yb + (long)row * E_DIM;
  for (int c = threadIdx.x; c < E_DIM; c += blockDim.x) {
    float v = (valid ? src[c] : 0.0f) + pe_val(t, c);
    dst[c] = v; dstb[c] = f2bf(v);
  }
}

// ---------------------------------------------------------------------------
// Host side
// ---------------------------------------------------------------------------
static inline void launch_gemm(hipStream_t st, int outbf,
                               const __bf16* A, long lda, long sA,
                               const __bf16* B, long ldb, long sB,
                               const float* bias, void* C, long ldc, long sC,
                               int M, int Nc, int Kd, int batch,
                               int relu, float scale)
{
  int waves = (M >> 4) * (Nc >> 6);
  dim3 grid((unsigned)((waves + 3) / 4), 1, (unsigned)batch);
  if (outbf)
    k_gemm<1><<<grid, 128, 0, st>>>(A, lda, sA, B, ldb, sB, bias, C, ldc, sC, M, Nc, Kd, relu, scale);
  else
    k_gemm<0><<<grid, 128, 0, st>>>(A, lda, sA, B, ldb, sB, bias, C, ldc, sC, M, Nc, Kd, relu, scale);
}

static inline void launch_gemm_big(hipStream_t st, int outbf,
                                   const __bf16* A, long lda,
                                   const __bf16* B, long ldb,
                                   const float* bias, void* C, long ldc,
                                   int M, int Nc, int Kd, int relu, float scale)
{
  dim3 grid((unsigned)((M >> 7) * (Nc >> 6)));
  if (outbf)
    k_gemm_big<1><<<grid, 128, 0, st>>>(A, lda, B, ldb, bias, C, ldc, M, Nc, Kd, relu, scale);
  else
    k_gemm_big<0><<<grid, 128, 0, st>>>(A, lda, B, ldb, bias, C, ldc, M, Nc, Kd, relu, scale);
}

extern "C" void kernel_launch(void* const* d_in, const int* in_sizes, int n_in,
                              void* d_out, int out_size, void* d_ws, size_t ws_size,
                              hipStream_t stream)
{
  (void)in_sizes; (void)n_in; (void)out_size; (void)ws_size;

  const int* inp = (const int*)d_in[0];
  const float* F[48];
  for (int i = 1; i < 48; ++i) F[i] = (const float*)d_in[i];
  const float *dW1=F[1], *dW2=F[2], *dWk=F[3], *dWk2=F[4], *dWo=F[5], *dWo2=F[6],
              *dWq=F[7], *dWq2=F[8], *dWv=F[9], *dWv2=F[10],
              *db1=F[11], *db2=F[12], *dbk=F[13], *dbk2=F[14], *dbo=F[15], *dbo2=F[16],
              *dbq=F[17], *dbq2=F[18], *dbv=F[19], *dbv2=F[20],
              *dln1b=F[21], *dln1g=F[22], *dln2b=F[23], *dln2g=F[24],
              *dln3b=F[25], *dln3g=F[26];
  const float* emb = F[27];
  const float *eW1=F[28], *eW2=F[29], *eWk=F[30], *eWo=F[31], *eWq=F[32], *eWv=F[33],
              *eb1=F[34], *eb2=F[35], *ebk=F[36], *ebo=F[37], *ebq=F[38], *ebv=F[39],
              *eln1b=F[40], *eln1g=F[41], *eln2b=F[42], *eln2g=F[43];
  const float *fertW=F[44], *fertb=F[45], *outW=F[46], *outb=F[47];

  float* w = (float*)d_ws;
  size_t off = 0;
  auto alloc  = [&](size_t n) { float* p = w + off; off += (n + 63) & ~(size_t)63; return p; };
  auto allocb = [&](size_t n) { return (__bf16*)alloc((n + 1) / 2); };
  float*  bufX   = alloc((size_t)4096 * 512);
  float*  bufT   = alloc((size_t)4096 * 512);
  float*  inputE = alloc((size_t)512 * 512);
  float*  fertL  = alloc((size_t)512 * 8);
  int*    fert   = (int*)alloc(512);
  int*    cum    = (int*)alloc(512);
  __bf16* Xb   = allocb((size_t)4096 * 512);
  __bf16* Mb   = allocb((size_t)512 * 512);
  __bf16* Qb   = allocb((size_t)4096 * 512);
  __bf16* Kb   = allocb((size_t)4096 * 512);
  __bf16* Vb   = allocb((size_t)4096 * 512);
  __bf16* Vt   = allocb((size_t)64 * 64 * 512);
  __bf16* Ob   = allocb((size_t)4096 * 512);
  __bf16* Hb   = allocb((size_t)4096 * 2048);
  __bf16* Sb   = allocb((size_t)64 * 512 * 512);
  __bf16* wbuf = allocb((size_t)512 * 32000);

  const long EE = (long)512 * 512;
  const long EF = (long)512 * 2048;
  const long NE = (long)NB * E_DIM;

  auto conv = [&](const float* W, int Kd, int Nc) {
    dim3 g((unsigned)(Kd / 32), (unsigned)(Nc / 32));
    k_wconv<<<g, 256, 0, stream>>>(W, wbuf, Kd, Nc);
  };
  auto wgemm = [&](const __bf16* A, const float* W, const float* bias, void* C,
                   int outbf, int M, int Nc, int Kd, int relu) {
    conv(W, Kd, Nc);
    launch_gemm_big(stream, outbf, A, Kd, wbuf, Kd, bias, C, Nc, M, Nc, Kd, relu, 1.f);
  };
  auto attention = [&](int Lq, int Lk) {
    k_vtrans<<<dim3((unsigned)(Lk / 32), 2, 64), 256, 0, stream>>>(Vb, Vt, Lk);
    launch_gemm(stream, 1, Qb, NE, 64, Kb, NE, 64, nullptr, Sb, Lk, (long)Lq * Lk,
                Lq, Lk, 64, 64, 0, 0.125f);
    k_softmax_bf<<<(unsigned)(64 * Lq), 256, 0, stream>>>(Sb, Lk);
    launch_gemm(stream, 1, Sb, Lk, (long)Lq * Lk, Vt, Lk, (long)64 * Lk, nullptr,
                Ob, NE, 64, Lq, 64, Lk, 64, 0, 1.f);
  };

  k_embed<<<512, 256, 0, stream>>>(inp, emb, inputE, bufX, Xb);

  for (int i = 0; i < NLAY; ++i) {
    wgemm(Xb, eWq + i*EE, ebq + i*512, Qb, 1, 512, 512, 512, 0);
    wgemm(Xb, eWk + i*EE, ebk + i*512, Kb, 1, 512, 512, 512, 0);
    wgemm(Xb, eWv + i*EE, ebv + i*512, Vb, 1, 512, 512, 512, 0);
    attention(64, 64);
    wgemm(Ob, eWo + i*EE, ebo + i*512, bufT, 0, 512, 512, 512, 0);
    k_add_ln<<<512, 256, 0, stream>>>(bufX, bufT, eln1g + i*512, eln1b + i*512, bufX, Xb);
    wgemm(Xb, eW1 + i*EF, eb1 + i*2048, Hb, 1, 512, 2048, 512, 1);
    wgemm(Hb, eW2 + i*EF, eb2 + i*512, bufT, 0, 512, 512, 2048, 0);
    k_add_ln<<<512, 256, 0, stream>>>(bufX, bufT, eln2g + i*512, eln2b + i*512, bufX, Xb);
  }

  k_fert_logits<<<512, 256, 0, stream>>>(bufX, fertW, fertb, fertL);
  k_argmax<<<2, 256, 0, stream>>>(fertL, fert);
  k_cumsum<<<NB, SEQ_S, 0, stream>>>(fert, cum);
  hipMemcpyAsync(Mb, Xb, (size_t)512 * 512 * sizeof(__bf16),
                 hipMemcpyDeviceToDevice, stream);
  k_copyfert<<<4096, 256, 0, stream>>>(inputE, cum, bufX, Xb);

  for (int i = 0; i < NLAY; ++i) {
    wgemm(Xb, dWq + i*EE, dbq + i*512, Qb, 1, 4096, 512, 512, 0);
    wgemm(Xb, dWk + i*EE, dbk + i*512, Kb, 1, 4096, 512, 512, 0);
    wgemm(Xb, dWv + i*EE, dbv + i*512, Vb, 1, 4096, 512, 512, 0);
    attention(512, 512);
    wgemm(Ob, dWo + i*EE, dbo + i*512, bufT, 0, 4096, 512, 512, 0);
    k_add_ln<<<4096, 256, 0, stream>>>(bufX, bufT, dln1g + i*512, dln1b + i*512, bufX, Xb);
    wgemm(Xb, dWq2 + i*EE, dbq2 + i*512, Qb, 1, 4096, 512, 512, 0);
    wgemm(Mb, dWk2 + i*EE, dbk2 + i*512, Kb, 1, 512, 512, 512, 0);
    wgemm(Mb, dWv2 + i*EE, dbv2 + i*512, Vb, 1, 512, 512, 512, 0);
    attention(512, 64);
    wgemm(Ob, dWo2 + i*EE, dbo2 + i*512, bufT, 0, 4096, 512, 512, 0);
    k_add_ln<<<4096, 256, 0, stream>>>(bufX, bufT, dln2g + i*512, dln2b + i*512, bufX, Xb);
    wgemm(Xb, dW1 + i*EF, db1 + i*2048, Hb, 1, 4096, 2048, 512, 1);
    wgemm(Hb, dW2 + i*EF, db2 + i*512, bufT, 0, 4096, 512, 2048, 0);
    k_add_ln<<<4096, 256, 0, stream>>>(bufX, bufT, dln3g + i*512, dln3b + i*512, bufX, Xb);
  }

  wgemm(Xb, outW, outb, d_out, 0, 4096, 32000, 512, 0);
}
